// TemporalAttentionBlock_6012954214530
// MI455X (gfx1250) — matrix operand order repaired and checked by reading, not compiled, not run on hardware
//
#include <hip/hip_runtime.h>
#include <hip/hip_bf16.h>
#include <math.h>

// Shapes (fixed by the reference)
#define BB   16
#define TT   256
#define VV   25
#define DD   128
#define HH   8
#define HD   16
#define MAXT 300
#define BIASW (2*MAXT-1)   // 599

typedef __bf16 bf16;
typedef __attribute__((ext_vector_type(16))) __bf16 v16bf;
typedef __attribute__((ext_vector_type(8)))  float  v8f;
typedef unsigned int u32x4 __attribute__((ext_vector_type(4)));
typedef int          i32x8 __attribute__((ext_vector_type(8)));
typedef int          i32x4 __attribute__((ext_vector_type(4)));

#if __has_builtin(__builtin_amdgcn_tensor_load_to_lds) && \
    __has_builtin(__builtin_amdgcn_s_wait_tensorcnt) && !defined(NO_TDM)
#define HAVE_TDM 1
#else
#define HAVE_TDM 0
#endif

union FragB {
    v16bf v;
    unsigned short h[16];
    unsigned u[8];
};

__device__ __forceinline__ v8f vzero8() {
    v8f z;
#pragma unroll
    for (int i = 0; i < 8; ++i) z[i] = 0.f;
    return z;
}

__device__ __forceinline__ v8f wmma_bf16(v16bf a, v16bf b, v8f c) {
    // D = A(16x32 bf16) * B(32x16 bf16) + C(16x16 f32)
    return __builtin_amdgcn_wmma_f32_16x16x32_bf16(false, a, false, b,
                                                   (short)0, c, false, false);
}

// LDS byte offset of a generic pointer that points into LDS
typedef __attribute__((address_space(3))) const char lds_cchar;
__device__ __forceinline__ unsigned lds_off_of(const void* p) {
    return (unsigned)(unsigned long long)(lds_cchar*)p;
}

#if HAVE_TDM
// Tensor Data Mover: DMA up-to-3D f32 tile from global to LDS.
// D# packing per CDNA5 ISA (data_size=4B).
__device__ __forceinline__ void tdm_load_f32(
    unsigned lds_off, const void* gptr,
    unsigned tile0, unsigned tile1, unsigned tile2,
    unsigned td0, unsigned td1, unsigned td2,
    unsigned long long s0, unsigned long long s1)
{
    unsigned long long ga = (unsigned long long)gptr;
    u32x4 g0;
    g0[0] = 1u;                                       // count=1, user mode
    g0[1] = lds_off;                                  // lds_addr (63:32)
    g0[2] = (unsigned)ga;                             // global_addr (95:64)
    g0[3] = (unsigned)((ga >> 32) & 0x1FFFFFFull) | 0x80000000u; // addr hi | type=2
    i32x8 g1;
    g1[0] = (int)(2u << 16);                          // workgroup_mask=0, data_size=4B
    g1[1] = (int)((td0 & 0xFFFFu) << 16);             // tensor_dim0 lo16 @ bit48
    g1[2] = (int)(((td0 >> 16) & 0xFFFFu) | ((td1 & 0xFFFFu) << 16));
    g1[3] = (int)(((td1 >> 16) & 0xFFFFu) | ((tile0 & 0xFFFFu) << 16));
    g1[4] = (int)((tile1 & 0xFFFFu) | ((tile2 & 0xFFFFu) << 16));
    g1[5] = (int)(s0 & 0xFFFFFFFFull);                // dim0 stride lo32
    g1[6] = (int)(((s0 >> 32) & 0xFFFFull) | ((s1 & 0xFFFFull) << 16));
    g1[7] = (int)((s1 >> 16) & 0xFFFFFFFFull);        // dim1 stride hi32
    i32x4 g2;
    g2[0] = (int)td2; g2[1] = 0; g2[2] = 0; g2[3] = 0;
    i32x4 g3;
    g3[0] = 0; g3[1] = 0; g3[2] = 0; g3[3] = 0;
#if defined(__clang_major__) && __clang_major__ >= 23
    i32x8 g4;
#pragma unroll
    for (int i = 0; i < 8; ++i) g4[i] = 0;
    __builtin_amdgcn_tensor_load_to_lds(g0, g1, g2, g3, g4, 0);
#else
    __builtin_amdgcn_tensor_load_to_lds(g0, g1, g2, g3, 0);
#endif
}
#endif

// ---- Fragment loaders (wave32; this file's contiguous k = 16*half + s in A and B, not the ISA table order) ----
// A (16-bit, 16x32): lanes 0-15 hold K=0..15, lanes 16-31 hold K=16..31
__device__ __forceinline__ v16bf load_A_lds(const bf16* m, int ld, int row0,
                                            int k0, int kmax, int lane) {
    const int half = (lane >> 4) & 1, mr = lane & 15;
    const bf16* rp = m + (size_t)(row0 + mr) * ld;
    FragB f;
#pragma unroll
    for (int i = 0; i < 8; ++i) {
        int k = k0 + half * 16 + (i << 1);
        f.u[i] = (k < kmax) ? *(const unsigned*)(rp + k) : 0u;  // pair k,k+1
    }
    return f.v;
}

// B (16-bit, 32x16): lanes 0-15 hold K=0..15, lanes 16-31 hold K=16..31.
// B[k][n] = m[k*ld + col0 + n]  (row-major KxN bf16 in LDS)
__device__ __forceinline__ v16bf load_B_lds(const bf16* m, int ld, int col0,
                                            int k0, int kmax, int lane) {
    const int half = (lane >> 4) & 1, mr = lane & 15;
    FragB f;
#pragma unroll
    for (int e = 0; e < 16; ++e) {
        int k = k0 + half * 16 + e;
        f.h[e] = (k < kmax)
                     ? *(const unsigned short*)(m + (size_t)k * ld + col0 + mr)
                     : (unsigned short)0;
    }
    return f.v;
}

// B where logical B[k][n] = m[(col0+n)*ld + k]  (K contiguous per row)
__device__ __forceinline__ v16bf load_BT_lds(const bf16* m, int ld, int col0,
                                             int k0, int kmax, int lane) {
    const int half = (lane >> 4) & 1, mr = lane & 15;
    const bf16* rp = m + (size_t)(col0 + mr) * ld;
    FragB f;
#pragma unroll
    for (int i = 0; i < 8; ++i) {
        int k = k0 + half * 16 + (i << 1);
        f.u[i] = (k < kmax) ? *(const unsigned*)(rp + k) : 0u;
    }
    return f.v;
}

// B from f32 row-major [K x ld] in LDS, cvt -> bf16 on the fly
__device__ __forceinline__ v16bf load_B_ldsf32(const float* m, int ld, int col0,
                                               int k0, int kmax, int lane) {
    const int half = (lane >> 4) & 1, mr = lane & 15;
    FragB f;
#pragma unroll
    for (int e = 0; e < 16; ++e) {
        int k = k0 + half * 16 + e;
        float x = (k < kmax) ? m[(size_t)k * ld + col0 + mr] : 0.f;
        bf16 bv = (bf16)x;
        f.h[e] = *(unsigned short*)&bv;
    }
    return f.v;
}

// 16-lane (half-wave) reductions
__device__ __forceinline__ float red16_sum(float v) {
    v += __shfl_xor(v, 1, 32); v += __shfl_xor(v, 2, 32);
    v += __shfl_xor(v, 4, 32); v += __shfl_xor(v, 8, 32);
    return v;
}
__device__ __forceinline__ float red16_max(float v) {
    v = fmaxf(v, __shfl_xor(v, 1, 32)); v = fmaxf(v, __shfl_xor(v, 2, 32));
    v = fmaxf(v, __shfl_xor(v, 4, 32)); v = fmaxf(v, __shfl_xor(v, 8, 32));
    return v;
}

__global__ __launch_bounds__(256, 1)
void fused_temporal_attn_block(
    const float* __restrict__ x,      const float* __restrict__ ln1g,
    const float* __restrict__ ln1b,   const float* __restrict__ qkvw,
    const float* __restrict__ qkvb,   const float* __restrict__ projw,
    const float* __restrict__ projb,  const float* __restrict__ lscale,
    const float* __restrict__ rpb,    const float* __restrict__ ln2g,
    const float* __restrict__ ln2b,   const float* __restrict__ w1,
    const float* __restrict__ b1,     const float* __restrict__ w2,
    const float* __restrict__ b2,     float* __restrict__ out)
{
    __shared__ __align__(16) char smem[262144];          // 256 KB / WG (<=320KB)
    bf16* XN = (bf16*)smem;                              // 256x128 bf16 (xn / yn)
    char* rest = smem + 65536;
    // phase-2 regions
    float* WSLF = (float*)(rest);                        // 128x48 f32 qkv slice
    float* PSLF = (float*)(rest + 24576);                // 16x128 f32 proj slice
    float* BIASF = (float*)(rest + 32768);               // 599 f32 bias row
    bf16* QN   = (bf16*)(rest + 36864);                  // 256x16
    bf16* KN   = (bf16*)(rest + 45056);
    bf16* VN   = (bf16*)(rest + 53248);
    bf16* OUTH = (bf16*)(rest + 61440);
    bf16* SCR  = (bf16*)(rest + 69632);                  // 8 waves x 16x32
    // phase-4 regions (overlap phase-2, separated by barriers)
    bf16*  TBLK = (bf16*)(rest);                         // 256x128 bf16
    float* W1F  = (float*)(rest + 65536);                // 128x128 f32
    float* W2F  = (float*)(rest + 131072);               // 128x128 f32

    const int tid  = threadIdx.x;
    const int lane = tid & 31;
    const int w    = tid >> 5;                // 8 waves
    const int half = (lane >> 4) & 1;
    const int mr   = lane & 15;

    const int seq = blockIdx.x;               // b*V + v
    const int bb  = seq / VV, vv = seq % VV;
    const float* xseq = x + ((size_t)bb * TT * VV + vv) * DD;
    const size_t xrow = (size_t)VV * DD;

    // ---------------- Phase 1: LN1 -> XN (bf16) -----------------------------
    {
        const float* xr = xseq + (size_t)tid * xrow;
        float s1 = 0.f, s2 = 0.f;
        for (int c = 0; c < DD; c += 4) {
            const float4 p = *(const float4*)(xr + c);
            s1 += p.x + p.y + p.z + p.w;
            s2 += p.x*p.x + p.y*p.y + p.z*p.z + p.w*p.w;
        }
        const float mean = s1 * (1.f/128.f);
        const float var  = s2 * (1.f/128.f) - mean*mean;
        const float rstd = rsqrtf(var + 1e-5f);
        for (int c = 0; c < DD; c += 4) {
            const float4 p  = *(const float4*)(xr + c);
            const float4 g4 = *(const float4*)(ln1g + c);
            const float4 b4 = *(const float4*)(ln1b + c);
            bf16* dst = XN + (size_t)tid * DD + c;
            dst[0] = (bf16)((p.x - mean)*rstd*g4.x + b4.x);
            dst[1] = (bf16)((p.y - mean)*rstd*g4.y + b4.y);
            dst[2] = (bf16)((p.z - mean)*rstd*g4.z + b4.z);
            dst[3] = (bf16)((p.w - mean)*rstd*g4.w + b4.w);
        }
    }

    v8f acc[16];                 // out@proj accumulators, later FFN2 accum
#pragma unroll
    for (int i = 0; i < 16; ++i) acc[i] = vzero8();
    __syncthreads();

    // ---------------- Phase 2: attention, head loop -------------------------
    for (int h = 0; h < HH; ++h) {
        // stage per-head weight slices + bias row into LDS
#if HAVE_TDM
        if (w == 0) {
            // qkv slice: 3D tile x=16 cols (contig), y=3 comps (stride 128),
            // z=128 rows (stride 384); dest = WSLF[row][comp*16+dd]
            tdm_load_f32(lds_off_of(WSLF), qkvw + h * 16,
                         16, 3, 128, 16, 3, 128, 128ull, 384ull);
            __builtin_amdgcn_s_wait_tensorcnt(0);
        } else if (w == 1) {
            tdm_load_f32(lds_off_of(PSLF), projw + (size_t)h * 16 * 128,
                         128, 16, 0, 128, 16, 0, 128ull, 0ull);
            __builtin_amdgcn_s_wait_tensorcnt(0);
        } else if (w == 2) {
            tdm_load_f32(lds_off_of(BIASF), rpb + (size_t)h * BIASW,
                         BIASW, 1, 0, BIASW, 1, 0, (unsigned long long)BIASW, 0ull);
            __builtin_amdgcn_s_wait_tensorcnt(0);
        }
#else
        for (int i = tid; i < 128 * 48; i += 256) {
            int k = i / 48, c = i % 48, comp = c / 16, dd2 = c % 16;
            WSLF[i] = qkvw[(size_t)k * 384 + comp * 128 + h * 16 + dd2];
        }
        for (int i = tid; i < 16 * 128; i += 256)
            PSLF[i] = projw[(size_t)h * 2048 + i];
        for (int i = tid; i < BIASW; i += 256)
            BIASF[i] = rpb[(size_t)h * BIASW + i];
#endif
        __syncthreads();

        const float scaleh =
            __expf(fminf(lscale[h], 4.6051702f)) * 0.25f;  // exp(min(ls,ln100))/sqrt(d)

        // q,k,v projections: M=256, N=16 each, K=128
        for (int rtl = 0; rtl < 2; ++rtl) {
            const int rt = w * 2 + rtl;
            for (int comp = 0; comp < 3; ++comp) {
                v8f dacc = vzero8();
#pragma unroll
                for (int kb = 0; kb < 4; ++kb) {
                    v16bf a = load_A_lds(XN, DD, rt * 16, kb * 32, DD, lane);
                    v16bf bf = load_B_ldsf32(WSLF, 48, comp * 16, kb * 32, DD, lane);
                    dacc = wmma_bf16(a, bf, dacc);
                }
                const float qb = qkvb[comp * 128 + h * 16 + mr];
#pragma unroll
                for (int r = 0; r < 8; ++r) dacc[r] += qb;
                if (comp < 2) {  // cosine-normalize q and k rows (d=16 = one tile)
#pragma unroll
                    for (int r = 0; r < 8; ++r) {
                        float ss  = red16_sum(dacc[r] * dacc[r]);
                        float inv = 1.f / fmaxf(sqrtf(ss), 1e-12f);
                        if (comp == 0) inv *= scaleh;
                        dacc[r] *= inv;
                    }
                }
                bf16* dst = (comp == 0) ? QN : ((comp == 1) ? KN : VN);
#pragma unroll
                for (int r = 0; r < 8; ++r)
                    dst[(size_t)(rt * 16 + r + 8 * half) * 16 + mr] = (bf16)dacc[r];
            }
        }
        __syncthreads();

        // scores + softmax + attn@V per row-tile (scores live in VGPRs)
        for (int rtl = 0; rtl < 2; ++rtl) {
            const int rt = w * 2 + rtl;
            v16bf aq = load_A_lds(QN, 16, rt * 16, 0, 16, lane);  // K=16 pad->32
            v8f sc[16];
#pragma unroll
            for (int ct = 0; ct < 16; ++ct) {
                v16bf bk = load_BT_lds(KN, 16, ct * 16, 0, 16, lane);
                sc[ct] = wmma_bf16(aq, bk, vzero8());
            }
            // relative position bias (from LDS)
#pragma unroll
            for (int ct = 0; ct < 16; ++ct) {
                const int s0 = ct * 16 + mr;
#pragma unroll
                for (int r = 0; r < 8; ++r) {
                    const int t0 = rt * 16 + r + 8 * half;
                    sc[ct][r] += BIASF[t0 - s0 + (MAXT - 1)];
                }
            }
            // row softmax (row spans 16 lanes of one half x 16 tiles)
#pragma unroll
            for (int r = 0; r < 8; ++r) {
                float mx = -3.4e38f;
#pragma unroll
                for (int ct = 0; ct < 16; ++ct) mx = fmaxf(mx, sc[ct][r]);
                mx = red16_max(mx);
                float sm = 0.f;
#pragma unroll
                for (int ct = 0; ct < 16; ++ct) {
                    float e = __expf(sc[ct][r] - mx);
                    sc[ct][r] = e;
                    sm += e;
                }
                sm = red16_sum(sm);
                const float inv = 1.f / sm;
#pragma unroll
                for (int ct = 0; ct < 16; ++ct) sc[ct][r] *= inv;
            }
            // out_h tile = attn(16x256) @ V(256x16), staged 32 cols at a time
            bf16* scw = SCR + w * 512;   // 16x32 bf16 per wave
            v8f oacc = vzero8();
            for (int kb = 0; kb < 8; ++kb) {
#pragma unroll
                for (int j = 0; j < 2; ++j) {
                    const int ct = kb * 2 + j;
#pragma unroll
                    for (int r = 0; r < 8; ++r)
                        scw[(r + 8 * half) * 32 + j * 16 + mr] = (bf16)sc[ct][r];
                }
                __syncthreads();
                v16bf aat = load_A_lds(scw, 32, 0, 0, 32, lane);
                v16bf bv  = load_B_lds(VN, 16, 0, kb * 32, 256, lane);
                oacc = wmma_bf16(aat, bv, oacc);
                __syncthreads();
            }
#pragma unroll
            for (int r = 0; r < 8; ++r)
                OUTH[(size_t)(rt * 16 + r + 8 * half) * 16 + mr] = (bf16)oacc[r];
        }
        __syncthreads();

        // accumulate out_h @ proj_w[h*16:+16, :] into persistent accumulators
        for (int rtl = 0; rtl < 2; ++rtl) {
            const int rt = w * 2 + rtl;
            v16bf ao = load_A_lds(OUTH, 16, rt * 16, 0, 16, lane);
#pragma unroll
            for (int ct = 0; ct < 8; ++ct) {
                v16bf bp = load_B_ldsf32(PSLF, 128, ct * 16, 0, 16, lane);
                acc[rtl * 8 + ct] = wmma_bf16(ao, bp, acc[rtl * 8 + ct]);
            }
        }
        __syncthreads();
    }

    // ------------- Phase 3: residual + LN2 (registers + shuffles) ----------
    v8f xfr[16];   // xf = x + out@proj + proj_b, kept in registers
#pragma unroll
    for (int rtl = 0; rtl < 2; ++rtl) {
#pragma unroll
        for (int ct = 0; ct < 8; ++ct) {
            const int idx = rtl * 8 + ct;
            const int col = ct * 16 + mr;
            const float pb = projb[col];
#pragma unroll
            for (int r = 0; r < 8; ++r) {
                const int row = w * 32 + rtl * 16 + r + 8 * half;
                xfr[idx][r] = acc[idx][r] + pb + xseq[(size_t)row * xrow + col];
                acc[idx][r] = 0.f;   // reuse for FFN2 accumulation
            }
        }
    }
#pragma unroll
    for (int rtl = 0; rtl < 2; ++rtl) {
#pragma unroll
        for (int r = 0; r < 8; ++r) {
            float s1 = 0.f, s2 = 0.f;
#pragma unroll
            for (int ct = 0; ct < 8; ++ct) {
                const float v2 = xfr[rtl * 8 + ct][r];
                s1 += v2; s2 += v2 * v2;
            }
            s1 = red16_sum(s1); s2 = red16_sum(s2);
            const float mean = s1 * (1.f/128.f);
            const float var  = s2 * (1.f/128.f) - mean * mean;
            const float rstd = rsqrtf(var + 1e-5f);
            const int row = w * 32 + rtl * 16 + r + 8 * half;
#pragma unroll
            for (int ct = 0; ct < 8; ++ct) {
                const int col = ct * 16 + mr;
                const float yv =
                    (xfr[rtl * 8 + ct][r] - mean) * rstd * ln2g[col] + ln2b[col];
                XN[(size_t)row * DD + col] = (bf16)yv;
            }
        }
    }
    __syncthreads();

    // ------------- Phase 4: FFN (4 blocks of 128 hidden cols) --------------
    for (int blk = 0; blk < 4; ++blk) {
#if HAVE_TDM
        if (w == 0) {
            tdm_load_f32(lds_off_of(W1F), w1 + blk * 128,
                         128, 128, 0, 128, 128, 0, 512ull, 0ull);
            __builtin_amdgcn_s_wait_tensorcnt(0);
        } else if (w == 1) {
            tdm_load_f32(lds_off_of(W2F), w2 + (size_t)blk * 128 * 128,
                         128, 128, 0, 128, 128, 0, 128ull, 0ull);
            __builtin_amdgcn_s_wait_tensorcnt(0);
        }
#else
        for (int i = tid; i < 128 * 128; i += 256) {
            int k = i >> 7, n = i & 127;
            W1F[i] = w1[(size_t)k * 512 + blk * 128 + n];
        }
        for (int i = tid; i < 128 * 128; i += 256)
            W2F[i] = w2[(size_t)blk * 16384 + i];
#endif
        __syncthreads();

        for (int rtl = 0; rtl < 2; ++rtl) {
            const int rt = w * 2 + rtl;
            v8f tacc[8];
#pragma unroll
            for (int ct = 0; ct < 8; ++ct) tacc[ct] = vzero8();
#pragma unroll
            for (int kb = 0; kb < 4; ++kb) {
                v16bf a = load_A_lds(XN, DD, rt * 16, kb * 32, DD, lane);
#pragma unroll
                for (int ct = 0; ct < 8; ++ct) {
                    v16bf bf = load_B_ldsf32(W1F, 128, ct * 16, kb * 32, DD, lane);
                    tacc[ct] = wmma_bf16(a, bf, tacc[ct]);
                }
            }
#pragma unroll
            for (int ct = 0; ct < 8; ++ct) {
                const int colg = blk * 128 + ct * 16 + mr;
                const float bb1 = b1[colg];
#pragma unroll
                for (int r = 0; r < 8; ++r) {
                    const float v2 = tacc[ct][r] + bb1;
                    const float g  = 0.5f * v2 * (1.f + erff(v2 * 0.70710678f));
                    TBLK[(size_t)(rt * 16 + r + 8 * half) * DD + ct * 16 + mr] =
                        (bf16)g;
                }
            }
        }
        __syncthreads();
        for (int rtl = 0; rtl < 2; ++rtl) {
            const int rt = w * 2 + rtl;
#pragma unroll
            for (int kb = 0; kb < 4; ++kb) {
                v16bf a = load_A_lds(TBLK, DD, rt * 16, kb * 32, DD, lane);
#pragma unroll
                for (int ct = 0; ct < 8; ++ct) {
                    v16bf bf = load_B_ldsf32(W2F, 128, ct * 16, kb * 32, 128, lane);
                    acc[rtl * 8 + ct] = wmma_bf16(a, bf, acc[rtl * 8 + ct]);
                }
            }
        }
        __syncthreads();
    }

    // ------------- Final: out = xf + ffn + b2, with (b,t,v,d) transpose ----
#pragma unroll
    for (int rtl = 0; rtl < 2; ++rtl) {
#pragma unroll
        for (int ct = 0; ct < 8; ++ct) {
            const int idx = rtl * 8 + ct;
            const int col = ct * 16 + mr;
            const float bb2 = b2[col];
#pragma unroll
            for (int r = 0; r < 8; ++r) {
                const int row = w * 32 + rtl * 16 + r + 8 * half;
                out[((size_t)(bb * TT + row) * VV + vv) * DD + col] =
                    xfr[idx][r] + acc[idx][r] + bb2;
            }
        }
    }
}

extern "C" void kernel_launch(void* const* d_in, const int* in_sizes, int n_in,
                              void* d_out, int out_size, void* d_ws, size_t ws_size,
                              hipStream_t stream) {
    const float* x      = (const float*)d_in[0];
    const float* ln1g   = (const float*)d_in[1];
    const float* ln1b   = (const float*)d_in[2];
    const float* qkvw   = (const float*)d_in[3];
    const float* qkvb   = (const float*)d_in[4];
    const float* projw  = (const float*)d_in[5];
    const float* projb  = (const float*)d_in[6];
    const float* lscale = (const float*)d_in[7];
    const float* rpb    = (const float*)d_in[8];
    const float* ln2g   = (const float*)d_in[9];
    const float* ln2b   = (const float*)d_in[10];
    const float* w1     = (const float*)d_in[11];
    const float* b1     = (const float*)d_in[12];
    const float* w2     = (const float*)d_in[13];
    const float* b2     = (const float*)d_in[14];
    float* out = (float*)d_out;

    dim3 grid(BB * VV), block(256);
    fused_temporal_attn_block<<<grid, block, 0, stream>>>(
        x, ln1g, ln1b, qkvw, qkvb, projw, projb, lscale, rpb,
        ln2g, ln2b, w1, b1, w2, b2, out);
}
